// ResidualVQ_24893630448083
// MI455X (gfx1250) — matrix operand order repaired and checked by reading, not compiled, not run on hardware
//
#include <hip/hip_runtime.h>
#include <hip/hip_bf16.h>
#include <float.h>

// Problem dims
#define BN      64
#define SN      256
#define DIM     512
#define NQ      8
#define KCB     1024
#define NROWS   16384          // BN*SN
#define QOUT_N  (NROWS*DIM)    // 8388608
#define IDX_OFF QOUT_N
#define IDX_N   (NROWS*NQ)     // 131072
#define LOSS_OFF (QOUT_N + IDX_N)

#define WG_ROWS 64             // rows per workgroup (4 waves x 16)
#define NTHREADS 128
#define STAGE_DW 8192          // 32 cols x 512 dims bf16 = 32KB = 8192 dwords

typedef __attribute__((ext_vector_type(16))) __bf16 v16bf;
typedef __attribute__((ext_vector_type(8)))  float  v8f;

#if __has_builtin(__builtin_amdgcn_tensor_load_to_lds)
#define USE_TDM 1
typedef unsigned u32x4 __attribute__((ext_vector_type(4)));
typedef int      i32x8 __attribute__((ext_vector_type(8)));
typedef int      i32x4 __attribute__((ext_vector_type(4)));
#endif

union BF16x16 {
    unsigned short us[16];
    v16bf          bf;
};

__device__ __forceinline__ unsigned short f2bf(float v) {
    unsigned u = __float_as_uint(v);
    unsigned r = (u + 0x7fffu + ((u >> 16) & 1u)) >> 16;   // round-to-nearest-even
    return (unsigned short)r;
}

#ifdef USE_TDM
// Issue one TDM load of 32KB (8192 dwords, contiguous) global -> LDS.
// D# per CDNA5 ISA ch.8: group0 = {count|flags, lds_addr, gaddr_lo, gaddr_hi|type},
// group1 = {mask|data_size, dims/tiles/strides}, groups 2/3 zero (<=2D tensor).
// This toolchain (clang-23 / therock-10.0) uses the 6-arg builtin form:
// (u32x4 g0, i32x8 g1, i32x4 g2, i32x4 g3, i32x8 extra, i32 cpol).
__device__ __forceinline__ void tdm_stage_32k(unsigned lds_off, const void* gptr) {
    unsigned long long ga = (unsigned long long)gptr;
    u32x4 g0;
    g0.x = 1u;                                              // count=1, user desc
    g0.y = lds_off;                                         // LDS byte address
    g0.z = (unsigned)ga;                                    // global_addr[31:0]
    g0.w = ((unsigned)(ga >> 32) & 0x01FFFFFFu) | (2u << 30); // [120:96] | type=2
    i32x8 g1;
    g1[0] = (int)(2u << 16);                 // workgroup_mask=0, data_size=2 (4B)
    g1[1] = (int)((STAGE_DW & 0xFFFFu) << 16);      // tensor_dim0[15:0]
    g1[2] = (int)((STAGE_DW >> 16) | (1u << 16));   // tensor_dim0[31:16] | tensor_dim1=1
    g1[3] = (int)((STAGE_DW & 0xFFFFu) << 16);      // tensor_dim1 hi=0 | tile_dim0
    g1[4] = 1;                                      // tile_dim1=1, tile_dim2=0
    g1[5] = (int)STAGE_DW;                          // tensor_dim0_stride[31:0]
    g1[6] = 0;
    g1[7] = 0;
    i32x4 z4 = {0, 0, 0, 0};
    i32x8 z8 = {0, 0, 0, 0, 0, 0, 0, 0};
    __builtin_amdgcn_tensor_load_to_lds(g0, g1, z4, z4, z8, 0);
}
#endif

// ---------------------------------------------------------------------------
// Kernel 1: convert codebooks to bf16 (into workspace) + precompute ||e||^2
// ---------------------------------------------------------------------------
__global__ __launch_bounds__(128) void rvq_prep_kernel(const float* __restrict__ embeds,
                                                       unsigned short* __restrict__ embB,
                                                       float* __restrict__ enorm) {
    const int code = blockIdx.x;                 // 0 .. NQ*KCB-1
    const float* src = embeds + (size_t)code * DIM;
    unsigned short* dst = embB + (size_t)code * DIM;
    float s = 0.0f;
    for (int j = threadIdx.x; j < DIM; j += 128) {
        float v = src[j];
        dst[j] = f2bf(v);
        s += v * v;
    }
    __shared__ float red[128];
    red[threadIdx.x] = s;
    __syncthreads();
    for (int off = 64; off > 0; off >>= 1) {
        if (threadIdx.x < off) red[threadIdx.x] += red[threadIdx.x + off];
        __syncthreads();
    }
    if (threadIdx.x == 0) enorm[code] = red[0];
}

// ---------------------------------------------------------------------------
// Kernel 2: residual VQ for a 64-row tile, all 8 quantizers in-block.
// 4 waves x 16 rows; B column-tiles staged into LDS by the Tensor Data Mover
// (double-buffered, 2 tiles per stage -> 2 independent WMMA acc chains).
// Grid = NROWS/64 = 256 blocks, 128 threads.
// ---------------------------------------------------------------------------
__global__ __launch_bounds__(NTHREADS) void rvq_main_kernel(const float* __restrict__ x,
                                                            const float* __restrict__ embeds,
                                                            const unsigned short* __restrict__ embB,
                                                            const float* __restrict__ enorm,
                                                            float* __restrict__ out) {
    // Big-LDS layout (CDNA5: up to 320KB per workgroup):
    __shared__ float          resF[WG_ROWS * DIM];          // 128 KB fp32 residual
    __shared__ unsigned short Bstage[2][32 * DIM];          // 2 x 32KB B staging
    __shared__ int            idxL[WG_ROWS];
    __shared__ float          lred[NTHREADS];

    const int tid  = threadIdx.x;
    const int lane = tid & 31;
    const int wave = tid >> 5;
    const int rowA = lane & 15;        // A-row / B-column / C-column within tile
    const int sel  = lane >> 4;        // lane-half select per ISA 16-bit layouts
    const size_t r0 = (size_t)blockIdx.x * WG_ROWS;

    // Load x tile -> fp32 residual in LDS
    for (int e = tid; e < WG_ROWS * DIM; e += NTHREADS)
        resF[e] = x[r0 * DIM + e];

    float tloss = 0.0f;

    for (int qp = 0; qp < NQ; ++qp) {
        __syncthreads();

        const unsigned short* EB = embB  + (size_t)qp * KCB * DIM;
        const float*          EN = enorm + (size_t)qp * KCB;

#ifdef USE_TDM
        // Kick off the first 32KB stage via TDM (DMA; EXEC-independent, once per wave)
        if (wave == 0)
            tdm_stage_32k((unsigned)(size_t)&Bstage[0][0], EB);
#endif

        // --- Build this wave's A fragments (16 rows x 512 dims, bf16) in regs.
        // Contiguous K order (matches B): lane-half sel=0 holds K {0..15},
        // sel=1 holds K {16..31} of each 32-K step.
        const float* arow = &resF[(wave * 16 + rowA) * DIM];
        v16bf afrag[16];
#pragma unroll
        for (int ks = 0; ks < DIM / 32; ++ks) {
            BF16x16 u;
#pragma unroll
            for (int i = 0; i < 8; ++i)
                u.us[i] = f2bf(arow[ks * 32 + sel * 16 + i]);
#pragma unroll
            for (int i = 0; i < 8; ++i)
                u.us[8 + i] = f2bf(arow[ks * 32 + sel * 16 + 8 + i]);
            afrag[ks] = u.bf;
        }

        float best[8];
        int   bidx[8];
#pragma unroll
        for (int v = 0; v < 8; ++v) { best[v] = FLT_MAX; bidx[v] = 0x7fffffff; }

#ifndef USE_TDM
        // Fallback staging of first pair (cooperative copy)
        {
            const uint4* src = (const uint4*)EB;
            uint4* dst = (uint4*)&Bstage[0][0];
            for (int j = tid; j < STAGE_DW / 4; j += NTHREADS) dst[j] = src[j];
        }
        __syncthreads();
#endif

        // --- Main loop over 32 pairs of column tiles (whole codebook)
        for (int p = 0; p < KCB / 32; ++p) {
            const int buf = p & 1;
#ifdef USE_TDM
            if (wave == 0) {
                if (p + 1 < KCB / 32) {
                    // Issue next stage, then wait for the OLDER one (TDM ops
                    // from one wave complete in-order; tensorcnt<=1 => p done).
                    tdm_stage_32k((unsigned)(size_t)&Bstage[buf ^ 1][0],
                                  EB + (size_t)(p + 1) * STAGE_DW * 2);
                    __builtin_amdgcn_s_wait_tensorcnt(1);
                } else {
                    __builtin_amdgcn_s_wait_tensorcnt(0);
                }
            }
            __syncthreads();   // publish Bstage[buf] to all waves
#else
            if (p + 1 < KCB / 32) {
                const uint4* src = (const uint4*)(EB + (size_t)(p + 1) * STAGE_DW * 2);
                uint4* dst = (uint4*)&Bstage[buf ^ 1][0];
                __builtin_prefetch(EB + (size_t)(p + 2) * STAGE_DW * 2, 0, 0);
                for (int j = tid; j < STAGE_DW / 4; j += NTHREADS) dst[j] = src[j];
            }
#endif

            v8f acc0 = {};
            v8f acc1 = {};
            const unsigned short* b0 = &Bstage[buf][(rowA)      * DIM + sel * 16];
            const unsigned short* b1 = &Bstage[buf][(16 + rowA) * DIM + sel * 16];
#pragma unroll
            for (int ks = 0; ks < DIM / 32; ++ks) {
                v16bf bm0 = *(const v16bf*)(b0 + ks * 32);
                v16bf bm1 = *(const v16bf*)(b1 + ks * 32);
                acc0 = __builtin_amdgcn_wmma_f32_16x16x32_bf16(
                    false, afrag[ks], false, bm0, (short)0, acc0, false, false);
                acc1 = __builtin_amdgcn_wmma_f32_16x16x32_bf16(
                    false, afrag[ks], false, bm1, (short)0, acc1, false, false);
            }
            const int c0 = p * 32 + rowA;
            const int c1 = c0 + 16;
            const float en0 = EN[c0];
            const float en1 = EN[c1];
#pragma unroll
            for (int v = 0; v < 8; ++v) {
                float d0 = en0 - 2.0f * acc0[v];       // ||e||^2 - 2 r.e
                float d1 = en1 - 2.0f * acc1[v];
                if (d0 < best[v]) { best[v] = d0; bidx[v] = c0; }
                if (d1 < best[v]) { best[v] = d1; bidx[v] = c1; }
            }
            __syncthreads();   // all reads of Bstage[buf] done before it is re-staged
        }

        // --- Cross-lane argmin within each 16-lane group (C layout: lane=col,
        // VGPR v = row sel*8+v). No cross-wave combine needed: wave owns rows.
#pragma unroll
        for (int m = 1; m < 16; m <<= 1) {
#pragma unroll
            for (int v = 0; v < 8; ++v) {
                float ob = __shfl_xor(best[v], m, 32);
                int   oi = __shfl_xor(bidx[v], m, 32);
                if (ob < best[v] || (ob == best[v] && oi < bidx[v])) {
                    best[v] = ob; bidx[v] = oi;
                }
            }
        }
        if (rowA == 0) {   // lanes 0 and 16 hold rows sel*8 .. sel*8+7
#pragma unroll
            for (int v = 0; v < 8; ++v) {
                const int wrow = wave * 16 + sel * 8 + v;
                idxL[wrow] = bidx[v];
                out[IDX_OFF + (r0 + (size_t)wrow) * NQ + qp] = (float)bidx[v];
            }
        }
        __syncthreads();

        // --- Residual update with fp32 codebook + loss accumulation
        const float* Eq = embeds + (size_t)qp * KCB * DIM;
        for (int e = tid; e < WG_ROWS * DIM; e += NTHREADS) {
            const int row = e >> 9;                 // DIM = 512
            const int dim = e & (DIM - 1);
            float qv = Eq[(size_t)idxL[row] * DIM + dim];
            float nr = resF[e] - qv;
            resF[e] = nr;
            tloss += nr * nr;    // == (sg(q)-r)^2 == (q-sg(r))^2 forward value
        }
    }
    __syncthreads();

    // qout = x - residual_final
    for (int e = tid; e < WG_ROWS * DIM; e += NTHREADS)
        out[r0 * DIM + e] = x[r0 * DIM + e] - resF[e];

    // loss = sum_q 2*mean(new_res^2), block-reduced then atomically accumulated
    lred[tid] = tloss * (2.0f / (float)QOUT_N);
    __syncthreads();
    for (int off = 64; off > 0; off >>= 1) {
        if (tid < off) lred[tid] += lred[tid + off];
        __syncthreads();
    }
    if (tid == 0) atomicAdd(out + LOSS_OFF, lred[0]);
}

// ---------------------------------------------------------------------------
extern "C" void kernel_launch(void* const* d_in, const int* in_sizes, int n_in,
                              void* d_out, int out_size, void* d_ws, size_t ws_size,
                              hipStream_t stream) {
    const float* x      = (const float*)d_in[0];   // [64,256,512]
    const float* embeds = (const float*)d_in[1];   // [8,1024,512]
    float* out = (float*)d_out;

    // Workspace: bf16 codebooks (8 MB) then code norms (32 KB)
    unsigned short* embB  = (unsigned short*)d_ws;
    float*          enorm = (float*)((char*)d_ws + (size_t)NQ * KCB * DIM * sizeof(unsigned short));

    // Zero the atomically-accumulated loss slot (graph-capture safe)
    (void)hipMemsetAsync(out + LOSS_OFF, 0, sizeof(float), stream);

    rvq_prep_kernel<<<NQ * KCB, 128, 0, stream>>>(embeds, embB, enorm);
    rvq_main_kernel<<<NROWS / WG_ROWS, NTHREADS, 0, stream>>>(x, embeds, embB, enorm, out);
}
